// Decoder_26645977104859
// MI455X (gfx1250) — matrix operand order repaired and checked by reading, not compiled, not run on hardware
//
#include <hip/hip_runtime.h>

// ---------------------------------------------------------------------------
// Types for CDNA5 WMMA (wave32): v_wmma_f32_16x16x32_bf16
// ---------------------------------------------------------------------------
typedef __attribute__((ext_vector_type(16))) __bf16 v16bf;
typedef __attribute__((ext_vector_type(8)))  float  v8f;
typedef __attribute__((ext_vector_type(4)))  unsigned int u32x4;

// exact parameter type of the async-LDS builtins (per hipcc diagnostic):
// "__attribute__((__vector_size__(4 * sizeof(int)))) int __device__ *"
typedef int gccv4i __attribute__((__vector_size__(4 * sizeof(int))));
typedef __attribute__((address_space(1))) gccv4i* gp1_t;
typedef __attribute__((address_space(3))) gccv4i* lp3_t;

union FragU { u32x4 q[2]; v16bf v; };

#define HAS_ASYNC_LDS (__has_builtin(__builtin_amdgcn_global_load_async_to_lds_b128) && \
                       __has_builtin(__builtin_amdgcn_s_wait_asynccnt))

__device__ __forceinline__ unsigned short f2bf(float x) {
    unsigned int u = __float_as_uint(x);
    unsigned int r = u + 0x7FFFu + ((u >> 16) & 1u);   // round-to-nearest-even
    return (unsigned short)(r >> 16);
}

__device__ __forceinline__ float sigmoidf_(float x) {
    return 1.0f / (1.0f + __expf(-x));
}

// Sizes
#define BB 128
#define SS 50
#define UU 512
#define ZZ 2048            // 4*U
#define HSTATE (BB*UU)     // 65536 floats per lstm

// ---------------------------------------------------------------------------
// One-time: swizzle U (recurrent) matrices [512,2048] fp32 -> bf16 fragments.
// Fragment f = nt*16 + kt covers N tile nt (16 cols), K chunk kt (32 rows).
// Lane l < 16: K = kt*32 + 0..15 (N = l);  lane >= 16: K = kt*32 + 16..31.
// Element j of a lane = K offset j  (pairs (2j,2j+1) land in VGPR j lo/hi).
// ---------------------------------------------------------------------------
__global__ __launch_bounds__(256)
void k_swizzleU(const float* __restrict__ Um, const float* __restrict__ Uy,
                const float* __restrict__ Uf, unsigned short* __restrict__ ubf)
{
    const int blk  = blockIdx.x;          // 3*2048 fragments
    const int lstm = blk >> 11;
    const int f    = blk & 2047;
    const int nt   = f >> 4;
    const int kt   = f & 15;
    const float* U = (lstm == 0) ? Um : (lstm == 1) ? Uy : Uf;
    unsigned short* dst = ubf + (size_t)lstm * (UU * ZZ) + (size_t)f * 512;
    for (int e = threadIdx.x; e < 512; e += 256) {
        const int l = e >> 4;
        const int j = e & 15;
        const int n = nt * 16 + (l & 15);
        const int k = kt * 32 + ((l & 16) ? 16 : 0) + j;
        dst[e] = f2bf(U[(size_t)k * ZZ + n]);
    }
}

// ---------------------------------------------------------------------------
// One-time: encW[lstm][b][col] = enc(b,:) @ W[:512, col]   (enc = state_h)
// ---------------------------------------------------------------------------
__global__ __launch_bounds__(256)
void k_encW(const float* __restrict__ h0, const float* __restrict__ Wm,
            const float* __restrict__ Wy, const float* __restrict__ Wf,
            float* __restrict__ encW)
{
    const int gid  = blockIdx.x * 256 + threadIdx.x;   // 3*128*2048
    const int lstm = gid >> 18;
    const int b    = (gid >> 11) & 127;
    const int col  = gid & 2047;
    const float* W  = (lstm == 0) ? Wm : (lstm == 1) ? Wy : Wf;
    const float* hr = h0 + b * UU;
    float acc = 0.f;
    for (int u = 0; u < UU; ++u) acc += hr[u] * W[(size_t)u * ZZ + col];
    encW[gid] = acc;
}

// ---------------------------------------------------------------------------
// One-time: replicate initial h/c state for the 3 LSTMs, seed bf16 h buffer
// ---------------------------------------------------------------------------
__global__ __launch_bounds__(256)
void k_init(const float* __restrict__ h0, const float* __restrict__ c0,
            float* __restrict__ h, float* __restrict__ c,
            unsigned short* __restrict__ hbf)
{
    const int gid = blockIdx.x * 256 + threadIdx.x;    // 65536
    const float hv = h0[gid], cv = c0[gid];
    const unsigned short hb = f2bf(hv);
    #pragma unroll
    for (int l = 0; l < 3; ++l) {
        h[l * HSTATE + gid]   = hv;
        c[l * HSTATE + gid]   = cv;
        hbf[l * HSTATE + gid] = hb;       // parity-0 read buffer for step 0
    }
}

// ---------------------------------------------------------------------------
// Per-step: condW[lstm][b][col] = cond_t(b,:) @ W[512:, col]
// ---------------------------------------------------------------------------
__global__ __launch_bounds__(256)
void k_condW(const float* __restrict__ mx, const float* __restrict__ my,
             const float* __restrict__ yy, const float* __restrict__ ff,
             const float* __restrict__ fadj,
             const float* __restrict__ Wm, const float* __restrict__ Wy,
             const float* __restrict__ Wf,
             float* __restrict__ condW, int s)
{
    const int gid  = blockIdx.x * 256 + threadIdx.x;   // 3*128*2048
    const int lstm = gid >> 18;
    const int b    = (gid >> 11) & 127;
    const int col  = gid & 2047;
    const float* W  = (lstm == 0) ? Wm : (lstm == 1) ? Wy : Wf;
    const float* Wt = W + (size_t)UU * ZZ;
    const int base = (b * SS + s) * 4;
    float acc = 0.f;
    if (lstm < 2) {
        const float* arrs[5] = {mx, my, yy, ff, fadj};
        #pragma unroll
        for (int a = 0; a < 5; ++a)
            #pragma unroll
            for (int e = 0; e < 4; ++e)
                acc += arrs[a][base + e] * Wt[(size_t)(a * 4 + e) * ZZ + col];
    } else {
        const float* arrs[2] = {ff, fadj};
        #pragma unroll
        for (int a = 0; a < 2; ++a)
            #pragma unroll
            for (int e = 0; e < 4; ++e)
                acc += arrs[a][base + e] * Wt[(size_t)(a * 4 + e) * ZZ + col];
    }
    condW[gid] = acc;
}

// ---------------------------------------------------------------------------
// Per-step LSTM cell: z = h @ U + encW + condW + bias  ->  activations.
// Grid: 96 blocks = 3 lstms x 32 hidden-unit tiles (16 units each).
// Block = 8 waves; wave w owns batch rows 16w..16w+15 and all 4 gate tiles.
// K loop: 16 chunks of 32. B fragments (4 gates x 1KB) double-buffered in
// LDS; staged with GLOBAL_LOAD_ASYNC_TO_LDS_B128 (ASYNCcnt) when available.
// One barrier per chunk: stores into buf p^1 at iter kt+1 are ordered after
// barrier kt, which already covers all waves' iter kt-1 reads of that buffer.
// ---------------------------------------------------------------------------
__global__ __launch_bounds__(256)
void k_cell(const unsigned short* __restrict__ ubf,
            const float* __restrict__ encW, const float* __restrict__ condW,
            const float* __restrict__ bm, const float* __restrict__ by,
            const float* __restrict__ bf,
            float* __restrict__ hst, float* __restrict__ cst,
            const unsigned short* __restrict__ hbf_r,
            unsigned short* __restrict__ hbf_w)
{
    const int lstm = blockIdx.x >> 5;
    const int ut   = blockIdx.x & 31;
    const int u0   = ut * 16;

    const unsigned short* U  = ubf   + (size_t)lstm * (UU * ZZ);
    const float* eW          = encW  + (size_t)lstm * (BB * ZZ);
    const float* cW          = condW + (size_t)lstm * (BB * ZZ);
    const float* bias        = (lstm == 0) ? bm : (lstm == 1) ? by : bf;
    float* h                 = hst + lstm * HSTATE;
    float* c                 = cst + lstm * HSTATE;
    const unsigned short* hr = hbf_r + lstm * HSTATE;
    unsigned short* hw       = hbf_w + lstm * HSTATE;

    __shared__ __align__(16) unsigned short ldsB[2][4 * 512];   // 2 x 4 KB

    const int tid  = threadIdx.x;
    const int wave = tid >> 5;
    const int lane = tid & 31;
    const int m0   = wave * 16;

    // cooperative staging role: gate g_of_t, 16B slice sub
    const int g_of_t = tid >> 6;
    const int sub    = tid & 63;
    const unsigned short* gsrc =
        U + (size_t)((g_of_t * 32 + ut) * 16) * 512 + sub * 8;  // + kt*512
    const int lofs = g_of_t * 512 + sub * 8;

    // A fragment addressing (16-bit A 16x32 layout)
    const int arow = m0 + (lane & 15);
    const int aoff = (lane & 16) ? 16 : 0;
    const unsigned short* hRow = hr + arow * UU + aoff;

    v8f acc[4] = {};

#if HAS_ASYNC_LDS
    // prologue: async-stage chunk 0 directly into LDS (no VGPR round trip)
    __builtin_amdgcn_global_load_async_to_lds_b128(
        (gp1_t)(uintptr_t)(gsrc),
        (lp3_t)(unsigned int)(uintptr_t)(&ldsB[0][lofs]), 0, 0);
#else
    u32x4 bq = *(const u32x4*)(gsrc);
#endif

    for (int kt = 0; kt < 16; ++kt) {
#if HAS_ASYNC_LDS
        __builtin_amdgcn_s_wait_asynccnt(0);   // own slice of chunk kt landed
        __syncthreads();                       // everyone's slice landed
        if (kt < 15)                           // safe after barrier kt
            __builtin_amdgcn_global_load_async_to_lds_b128(
                (gp1_t)(uintptr_t)(gsrc + (size_t)(kt + 1) * 512),
                (lp3_t)(unsigned int)(uintptr_t)(&ldsB[(kt + 1) & 1][lofs]),
                0, 0);
#else
        *(u32x4*)(&ldsB[kt & 1][lofs]) = bq;
        __syncthreads();
        if (kt < 15) {
            bq = *(const u32x4*)(gsrc + (size_t)(kt + 1) * 512);
            __builtin_prefetch(gsrc + (size_t)(kt + 1) * 512, 0, 1);
        }
#endif
        // A fragment (two clause'd b128 loads from bf16 h state)
        FragU a;
        a.q[0] = *(const u32x4*)(hRow + kt * 32);
        a.q[1] = *(const u32x4*)(hRow + kt * 32 + 8);

        // all four gate B fragments first (8 ds_load_b128, one wait),
        // then 4 back-to-back WMMAs
        const unsigned short* lb = ldsB[kt & 1];
        FragU b0, b1, b2, b3;
        b0.q[0] = *(const u32x4*)(&lb[0 * 512 + lane * 16]);
        b0.q[1] = *(const u32x4*)(&lb[0 * 512 + lane * 16 + 8]);
        b1.q[0] = *(const u32x4*)(&lb[1 * 512 + lane * 16]);
        b1.q[1] = *(const u32x4*)(&lb[1 * 512 + lane * 16 + 8]);
        b2.q[0] = *(const u32x4*)(&lb[2 * 512 + lane * 16]);
        b2.q[1] = *(const u32x4*)(&lb[2 * 512 + lane * 16 + 8]);
        b3.q[0] = *(const u32x4*)(&lb[3 * 512 + lane * 16]);
        b3.q[1] = *(const u32x4*)(&lb[3 * 512 + lane * 16 + 8]);

        acc[0] = __builtin_amdgcn_wmma_f32_16x16x32_bf16(
                     false, a.v, false, b0.v, (short)0, acc[0], false, false);
        acc[1] = __builtin_amdgcn_wmma_f32_16x16x32_bf16(
                     false, a.v, false, b1.v, (short)0, acc[1], false, false);
        acc[2] = __builtin_amdgcn_wmma_f32_16x16x32_bf16(
                     false, a.v, false, b2.v, (short)0, acc[2], false, false);
        acc[3] = __builtin_amdgcn_wmma_f32_16x16x32_bf16(
                     false, a.v, false, b3.v, (short)0, acc[3], false, false);
    }

    // epilogue: C layout -> row = m0 + r + (lane&16 ? 8 : 0), col = u0 + lane&15
    const int uCol  = u0 + (lane & 15);
    const int rbase = m0 + ((lane & 16) ? 8 : 0);
    #pragma unroll
    for (int r = 0; r < 8; ++r) {
        const int bidx = rbase + r;
        float z[4];
        #pragma unroll
        for (int g = 0; g < 4; ++g) {
            const int col = g * 512 + uCol;
            z[g] = acc[g][r] + eW[bidx * ZZ + col] + cW[bidx * ZZ + col] + bias[col];
        }
        const float cold = c[bidx * UU + uCol];
        const float is = sigmoidf_(z[0]);
        const float fs = sigmoidf_(z[1]);
        const float gt = tanhf(z[2]);
        const float os = sigmoidf_(z[3]);
        const float cn = fs * cold + is * gt;
        const float hn = os * tanhf(cn);
        c[bidx * UU + uCol]  = cn;
        h[bidx * UU + uCol]  = hn;
        hw[bidx * UU + uCol] = f2bf(hn);
    }
}

// ---------------------------------------------------------------------------
// Per-step heads: 75 length-512 dots per batch element + softmax/exp/tanh.
// One block per batch element; h rows staged in LDS.
// ---------------------------------------------------------------------------
__global__ __launch_bounds__(128)
void k_heads(const float* __restrict__ hst,
             const float* __restrict__ hWm, const float* __restrict__ hbm,
             const float* __restrict__ hWy, const float* __restrict__ hby,
             const float* __restrict__ hWf, const float* __restrict__ hbff,
             const float* __restrict__ hWfa, const float* __restrict__ hbfa,
             float* __restrict__ out, int s)
{
    const int b = blockIdx.x;
    __shared__ float sh[3][UU];
    __shared__ float raw[75];
    for (int i = threadIdx.x; i < UU; i += 128) {
        sh[0][i] = hst[0 * HSTATE + b * UU + i];
        sh[1][i] = hst[1 * HSTATE + b * UU + i];
        sh[2][i] = hst[2 * HSTATE + b * UU + i];
    }
    __syncthreads();

    const int t = threadIdx.x;
    if (t < 75) {
        const int region = (t < 30) ? 0 : (t < 45) ? 1 : (t < 60) ? 2 : 3;
        const int rb = (region == 0) ? 0 : (region == 1) ? 30 : (region == 2) ? 45 : 60;
        const int loc = t - rb;
        const int n = loc / 5, k = loc % 5;
        const float* W  = (region == 0) ? hWm : (region == 1) ? hWy :
                          (region == 2) ? hWf : hWfa;
        const float* bb = (region == 0) ? hbm : (region == 1) ? hby :
                          (region == 2) ? hbff : hbfa;
        const float* hv = (region == 0) ? sh[0] : (region == 1) ? sh[1] : sh[2];
        float acc = bb[n * 5 + k];
        for (int u = 0; u < UU; ++u) acc += hv[u] * W[(size_t)(n * UU + u) * 5 + k];
        raw[t] = acc;
    }
    __syncthreads();
    if (t < 75) {
        const int region = (t < 30) ? 0 : (t < 45) ? 1 : (t < 60) ? 2 : 3;
        const int rb = (region == 0) ? 0 : (region == 1) ? 30 : (region == 2) ? 45 : 60;
        const int loc = t - rb;
        const int n = loc / 5;
        const float v = raw[t];
        float res;
        if (n == 0) {                       // softmax over this head's alpha row
            float mx = raw[rb];
            #pragma unroll
            for (int q = 1; q < 5; ++q) mx = fmaxf(mx, raw[rb + q]);
            float sum = 0.f;
            #pragma unroll
            for (int q = 0; q < 5; ++q) sum += expf(raw[rb + q] - mx);
            res = expf(v - mx) / sum;
        } else if ((region == 0 && (n == 2 || n == 4)) || (region != 0 && n == 2)) {
            res = expf(v);                  // sigmas
        } else if (region == 0 && n == 5) {
            res = tanhf(v);                 // rhos
        } else {
            res = v;                        // mus
        }
        float* op;
        if (region == 0) op = out + ((size_t)b * SS + s) * 30 + loc;
        else op = out + 192000 + (size_t)(region - 1) * 96000 + ((size_t)b * SS + s) * 15 + loc;
        *op = res;
    }
}

// ---------------------------------------------------------------------------
// Launch
// ---------------------------------------------------------------------------
extern "C" void kernel_launch(void* const* d_in, const int* in_sizes, int n_in,
                              void* d_out, int out_size, void* d_ws, size_t ws_size,
                              hipStream_t stream) {
    const float* cond_mx   = (const float*)d_in[0];
    const float* cond_my   = (const float*)d_in[1];
    const float* cond_y    = (const float*)d_in[2];
    const float* cond_f    = (const float*)d_in[3];
    const float* cond_fadj = (const float*)d_in[4];
    const float* state_h   = (const float*)d_in[5];
    const float* state_c   = (const float*)d_in[6];
    const float* Wm = (const float*)d_in[7];
    const float* Um = (const float*)d_in[8];
    const float* bm = (const float*)d_in[9];
    const float* Wy = (const float*)d_in[10];
    const float* Uy = (const float*)d_in[11];
    const float* by = (const float*)d_in[12];
    const float* Wf = (const float*)d_in[13];
    const float* Uf = (const float*)d_in[14];
    const float* bf = (const float*)d_in[15];
    const float* headW_m    = (const float*)d_in[16];
    const float* headb_m    = (const float*)d_in[17];
    const float* headW_y    = (const float*)d_in[18];
    const float* headb_y    = (const float*)d_in[19];
    const float* headW_f    = (const float*)d_in[20];
    const float* headb_f    = (const float*)d_in[21];
    const float* headW_fadj = (const float*)d_in[22];
    const float* headb_fadj = (const float*)d_in[23];
    float* out = (float*)d_out;

    char* ws = (char*)d_ws;
    unsigned short* ubf   = (unsigned short*)(ws);                        // 6 MB
    float*          encW  = (float*)(ws + (size_t)6  * 1024 * 1024);      // 3 MB
    float*          condW = (float*)(ws + (size_t)9  * 1024 * 1024);      // 3 MB
    float*          hst   = (float*)(ws + (size_t)12 * 1024 * 1024);      // 768 KB
    float*          cst   = (float*)(ws + (size_t)12 * 1024 * 1024 + 786432);
    unsigned short* hbf   = (unsigned short*)(ws + (size_t)12 * 1024 * 1024 + 2 * 786432);

    k_swizzleU<<<6144, 256, 0, stream>>>(Um, Uy, Uf, ubf);
    k_encW<<<3072, 256, 0, stream>>>(state_h, Wm, Wy, Wf, encW);
    k_init<<<256, 256, 0, stream>>>(state_h, state_c, hst, cst, hbf);

    for (int s = 0; s < SS; ++s) {
        const int pr = s & 1;
        k_condW<<<3072, 256, 0, stream>>>(cond_mx, cond_my, cond_y, cond_f, cond_fadj,
                                          Wm, Wy, Wf, condW, s);
        k_cell<<<96, 256, 0, stream>>>(ubf, encW, condW, bm, by, bf, hst, cst,
                                       hbf + (size_t)pr * (3 * HSTATE),
                                       hbf + (size_t)(1 - pr) * (3 * HSTATE));
        k_heads<<<128, 128, 0, stream>>>(hst, headW_m, headb_m, headW_y, headb_y,
                                         headW_f, headb_f, headW_fadj, headb_fadj,
                                         out, s);
    }
}
